// PSTTransOp_29600914604802
// MI455X (gfx1250) — matrix operand order repaired and checked by reading, not compiled, not run on hardware
//
#include <hip/hip_runtime.h>
#include <hip/hip_bf16.h>

// ---------------------------------------------------------------------------
// MI455X (gfx1250) implementation. Memory-bound workload (~21 GFLOP vs
// ~350 MB HBM traffic @ 23.3 TB/s), so: fuse both temporal GEMMs in LDS,
// keep intermediates bf16, use v_wmma_f32_16x16x32_bf16 (f32 accumulate)
// for all matrix ops, fuse BN+ReLU epilogue into the output GEMM.
// ---------------------------------------------------------------------------

typedef __attribute__((ext_vector_type(16))) __bf16 v16bf;
typedef __attribute__((ext_vector_type(8)))  float  v8f;

#define DEV_INLINE __device__ __forceinline__

static constexpr int B_    = 2;
static constexpr int L2_   = 4;
static constexpr int N2_   = 2048;
static constexpr int L1_   = 9;
static constexpr int N1_   = 8192;
static constexpr int CIN_  = 128;
static constexpr int CT_   = 128;
static constexpr int CS_   = 256;
static constexpr int CORIG_= 64;
static constexpr int KT_   = 3;              // temporal kernel taps
static constexpr int CCAT_ = CT_ + CORIG_;   // 192

typedef unsigned short bf16_t;               // raw bf16 storage bits

DEV_INLINE bf16_t f2bf(float f) {            // round-to-nearest-even
  unsigned u = __builtin_bit_cast(unsigned, f);
  u += 0x7FFFu + ((u >> 16) & 1u);
  return (bf16_t)(u >> 16);
}
DEV_INLINE float bf2f(bf16_t h) {
  unsigned u = ((unsigned)h) << 16;
  return __builtin_bit_cast(float, u);
}
DEV_INLINE __bf16 bits2bf(bf16_t h) { return __builtin_bit_cast(__bf16, h); }

// A fragment: 16x32 bf16 tile, rows rowBase..+15, cols kBase..+31 of a
// row-major LDS matrix with pitch P. Lane half 0 holds K 0..15, lane half 1
// holds K 16..31 (same k order as B fragment); VGPR v packs 2 Ks.
DEV_INLINE v16bf load_frag_a(const bf16_t* W, int P, int rowBase, int kBase,
                             int lane) {
  const int half = lane >> 4;
  const int m    = lane & 15;
  const bf16_t* rp = W + (rowBase + m) * P + kBase;
  v16bf a;
#pragma unroll
  for (int vp = 0; vp < 8; ++vp) {
    int kk = (half << 4) + (vp << 1);
    a[2 * vp]     = bits2bf(rp[kk]);
    a[2 * vp + 1] = bits2bf(rp[kk + 1]);
  }
  return a;
}

// B fragment: 32x16 bf16 tile B[k][n] sourced from LDS X[n][k] (row-major,
// pitch P). Contiguous k order (same as A): lanes 0-15 hold K=0..15 for column N=lane, lanes
// 16-31 hold K=16..31. -> 16 contiguous bf16 per lane (vectorizable ds load).
DEV_INLINE v16bf load_frag_b(const bf16_t* X, int P, int nBase, int kBase,
                             int lane) {
  const int half = lane >> 4;
  const int n    = lane & 15;
  const bf16_t* rp = X + (nBase + n) * P + kBase + (half << 4);
  v16bf b;
#pragma unroll
  for (int e = 0; e < 16; ++e) b[e] = bits2bf(rp[e]);
  return b;
}

// ---------------------------------------------------------------------------
// Kernel 1: tt = relu(Wt1 @ relu(Wt0 @ features)), fused in LDS.
// Grid: (b, l2, kt, n-tile of 64). Block 256 = 8 waves; wave w owns output
// rows [16w, 16w+16) x 64 columns (4 WMMA tiles).
// ---------------------------------------------------------------------------
__global__ __launch_bounds__(256) void k_temporal_conv(
    const float* __restrict__ features, const float* __restrict__ Wt0,
    const float* __restrict__ Wt1, bf16_t* __restrict__ tt) {
  __shared__ bf16_t sW[CT_ * CIN_];   // weights [o][c]          32 KB
  __shared__ bf16_t sX[64 * CIN_];    // activations [n][c]      16 KB

  const int tid  = threadIdx.x;
  const int lane = tid & 31;
  const int wave = tid >> 5;

  int bid = blockIdx.x;
  const int NTB = N2_ / 64;
  const int nt = bid % NTB; bid /= NTB;
  const int kt = bid % KT_; bid /= KT_;
  const int l2 = bid % L2_; bid /= L2_;
  const int b  = bid;
  const int nBase = nt * 64;

  // features (b,l2,c,n) f32 -> sX[n][c] bf16 (transposed stage)
  const float* fbase =
      features + (size_t)(b * L2_ + l2) * CIN_ * N2_ + nBase;
  for (int i = tid; i < 64 * CIN_; i += 256) {
    int c = i >> 6, n = i & 63;
    sX[n * CIN_ + c] = f2bf(fbase[(size_t)c * N2_ + n]);
  }
  // Wt0[kt] f32 -> sW bf16
  const float* w0 = Wt0 + (size_t)kt * CT_ * CIN_;
  for (int i = tid; i < CT_ * CIN_; i += 256) sW[i] = f2bf(w0[i]);
  __syncthreads();

  const v8f vzero = {0.f, 0.f, 0.f, 0.f, 0.f, 0.f, 0.f, 0.f};
  const int oBase = wave * 16;
  const int n0 = lane & 15;
  const int half = lane >> 4;

  // ---- stage 1: h = relu(Wt0 @ f) ----
  v8f acc[4];
#pragma unroll
  for (int t = 0; t < 4; ++t) acc[t] = vzero;
  for (int kb = 0; kb < CIN_; kb += 32) {
    v16bf afr = load_frag_a(sW, CIN_, oBase, kb, lane);
#pragma unroll
    for (int t = 0; t < 4; ++t) {
      v16bf bfr = load_frag_b(sX, CIN_, t * 16, kb, lane);
      acc[t] = __builtin_amdgcn_wmma_f32_16x16x32_bf16(
          false, afr, false, bfr, (short)0, acc[t], false, false);
    }
  }
  __syncthreads();               // everyone done reading sX
  // write relu(h) transposed back into sX[n][c]
#pragma unroll
  for (int t = 0; t < 4; ++t)
#pragma unroll
    for (int r = 0; r < 8; ++r) {
      float v = acc[t][r];
      v = v > 0.f ? v : 0.f;
      sX[(t * 16 + n0) * CIN_ + (oBase + r + 8 * half)] = f2bf(v);
    }
  __syncthreads();               // h visible, sW no longer read
  const float* w1 = Wt1 + (size_t)kt * CT_ * CT_;
  for (int i = tid; i < CT_ * CT_; i += 256) sW[i] = f2bf(w1[i]);
  __syncthreads();

  // ---- stage 2: tt = relu(Wt1 @ h) ----
#pragma unroll
  for (int t = 0; t < 4; ++t) acc[t] = vzero;
  for (int kb = 0; kb < CT_; kb += 32) {
    v16bf afr = load_frag_a(sW, CT_, oBase, kb, lane);
#pragma unroll
    for (int t = 0; t < 4; ++t) {
      v16bf bfr = load_frag_b(sX, CT_, t * 16, kb, lane);
      acc[t] = __builtin_amdgcn_wmma_f32_16x16x32_bf16(
          false, afr, false, bfr, (short)0, acc[t], false, false);
    }
  }
  bf16_t* outp =
      tt + (size_t)((b * L2_ + l2) * KT_ + kt) * CT_ * N2_ + nBase;
#pragma unroll
  for (int t = 0; t < 4; ++t)
#pragma unroll
    for (int r = 0; r < 8; ++r) {
      float v = acc[t][r];
      v = v > 0.f ? v : 0.f;
      outp[(size_t)(oBase + r + 8 * half) * N2_ + t * 16 + n0] = f2bf(v);
    }
}

// ---------------------------------------------------------------------------
// Kernel 2: 3-NN inverse-distance interpolation; builds bf16 x = [interp(128);
// original_features(64)] with layout (b, t1, c, n1).
// Grid: (b, t1, 256-point chunk). One lane per query point.
// ---------------------------------------------------------------------------
__global__ __launch_bounds__(256) void k_knn_interp(
    const float* __restrict__ xyzs, const float* __restrict__ original_xyzs,
    const float* __restrict__ original_features,
    const bf16_t* __restrict__ tt, bf16_t* __restrict__ xbuf) {
  __shared__ float sx[2 * N2_];
  __shared__ float sy[2 * N2_];
  __shared__ float sz[2 * N2_];

  const int tid = threadIdx.x;
  int bid = blockIdx.x;
  const int CH = N1_ / 256;
  const int chunk = bid % CH; bid /= CH;
  const int t1i = bid % L1_; bid /= L1_;
  const int b = bid;
  const int t1 = t1i + 1;

  // seed (t2, tap) list: middles m = 2*t2+2; include when |t1-m| <= R(=1)
  int st2[2], soff[2], S = 0;
#pragma unroll
  for (int t2 = 0; t2 < L2_; ++t2) {
    int m = 2 * t2 + 2;
    if (t1 >= m - 1 && t1 <= m + 1) { st2[S] = t2; soff[S] = t1 - m + 1; ++S; }
  }
  const int NS = S * N2_;

  for (int i = tid; i < NS; i += 256) {
    int s = i >> 11, n2 = i & (N2_ - 1);
    const float* p = xyzs + ((size_t)(b * L2_ + st2[s]) * N2_ + n2) * 3;
    sx[i] = p[0]; sy[i] = p[1]; sz[i] = p[2];
  }
  __syncthreads();

  const int n1 = chunk * 256 + tid;
  const float* ap =
      original_xyzs + ((size_t)(b * L1_ + t1i) * N1_ + n1) * 3;
  const float ax = ap[0], ay = ap[1], az = ap[2];

  float d0 = 3.4e38f, d1 = 3.4e38f, d2 = 3.4e38f;
  int i0 = 0, i1 = 0, i2 = 0;
  for (int i = 0; i < NS; ++i) {
    float dx = ax - sx[i], dy = ay - sy[i], dz = az - sz[i];
    float d = dx * dx + dy * dy + dz * dz;
    if (d < d0)      { d2 = d1; i2 = i1; d1 = d0; i1 = i0; d0 = d; i0 = i; }
    else if (d < d1) { d2 = d1; i2 = i1; d1 = d;  i1 = i; }
    else if (d < d2) { d2 = d;  i2 = i; }
  }
  float w0 = 1.f / (d0 + 1e-8f);
  float w1 = 1.f / (d1 + 1e-8f);
  float w2 = 1.f / (d2 + 1e-8f);
  const float wnorm = 1.f / (w0 + w1 + w2);
  w0 *= wnorm; w1 *= wnorm; w2 *= wnorm;

  size_t base[3];
  const int idx[3] = {i0, i1, i2};
  const float w[3] = {w0, w1, w2};
#pragma unroll
  for (int j = 0; j < 3; ++j) {
    int s = idx[j] >> 11, n2 = idx[j] & (N2_ - 1);
    base[j] =
        (size_t)((b * L2_ + st2[s]) * KT_ + soff[s]) * CT_ * N2_ + n2;
  }

  bf16_t* xb = xbuf + (size_t)(b * L1_ + t1i) * CCAT_ * N1_ + n1;
  for (int c = 0; c < CT_; ++c) {
    float g = w[0] * bf2f(tt[base[0] + (size_t)c * N2_]) +
              w[1] * bf2f(tt[base[1] + (size_t)c * N2_]) +
              w[2] * bf2f(tt[base[2] + (size_t)c * N2_]);
    xb[(size_t)c * N1_] = f2bf(g);
  }
  const float* of =
      original_features + (size_t)(b * L1_ + t1i) * CORIG_ * N1_ + n1;
  for (int c = 0; c < CORIG_; ++c)
    xb[(size_t)(CT_ + c) * N1_] = f2bf(of[(size_t)c * N1_]);
}

// ---------------------------------------------------------------------------
// Kernel 3: out = relu(bn_scale * (Ws @ x) + bn_bias), WMMA bf16, K=192.
// Grid: (b, t1, 64-row o-tile, 64-col n-tile). 8 waves: wave = 4*nHalf+oStrip.
// ---------------------------------------------------------------------------
__global__ __launch_bounds__(256) void k_out_gemm(
    const bf16_t* __restrict__ xbuf, const float* __restrict__ Ws,
    const float* __restrict__ bn_gamma, const float* __restrict__ bn_beta,
    const float* __restrict__ bn_mean, const float* __restrict__ bn_var,
    float* __restrict__ out) {
  __shared__ bf16_t sW[64 * CCAT_];   // Ws rows [o][c]   24 KB
  __shared__ bf16_t sX[64 * CCAT_];   // x tile  [n][c]   24 KB

  const int tid  = threadIdx.x;
  const int lane = tid & 31;
  const int wave = tid >> 5;

  int bid = blockIdx.x;
  const int NTB = N1_ / 64;
  const int nt = bid % NTB; bid /= NTB;
  const int ot = bid % (CS_ / 64); bid /= (CS_ / 64);
  const int t1i = bid % L1_; bid /= L1_;
  const int b = bid;
  const int nTileBase = nt * 64;
  const int oTileBase = ot * 64;

  for (int i = tid; i < 64 * CCAT_; i += 256)
    sW[i] = f2bf(Ws[(size_t)(oTileBase + i / CCAT_) * CCAT_ + (i % CCAT_)]);
  const bf16_t* xg =
      xbuf + (size_t)(b * L1_ + t1i) * CCAT_ * N1_ + nTileBase;
  for (int i = tid; i < 64 * CCAT_; i += 256) {
    int c = i >> 6, n = i & 63;
    sX[n * CCAT_ + c] = xg[(size_t)c * N1_ + n];
  }
  __syncthreads();

  const v8f vzero = {0.f, 0.f, 0.f, 0.f, 0.f, 0.f, 0.f, 0.f};
  const int oBase = (wave & 3) * 16;
  const int nHalf = (wave >> 2) * 32;

  v8f acc[2] = {vzero, vzero};
  for (int kb = 0; kb < CCAT_; kb += 32) {
    v16bf afr = load_frag_a(sW, CCAT_, oBase, kb, lane);
#pragma unroll
    for (int t = 0; t < 2; ++t) {
      v16bf bfr = load_frag_b(sX, CCAT_, nHalf + t * 16, kb, lane);
      acc[t] = __builtin_amdgcn_wmma_f32_16x16x32_bf16(
          false, afr, false, bfr, (short)0, acc[t], false, false);
    }
  }

  const int n0 = lane & 15;
  const int half = lane >> 4;
  const size_t xyzOff = (size_t)B_ * L1_ * N1_ * 3;   // 442368 floats
  float* ob =
      out + xyzOff + (size_t)(b * L1_ + t1i) * CS_ * N1_ + nTileBase;
#pragma unroll
  for (int r = 0; r < 8; ++r) {
    const int o = oTileBase + oBase + r + 8 * half;
    const float sc = bn_gamma[o] * rsqrtf(bn_var[o] + 1e-5f);
    const float bi = bn_beta[o] - bn_mean[o] * sc;
#pragma unroll
    for (int t = 0; t < 2; ++t) {
      float v = acc[t][r] * sc + bi;
      v = v > 0.f ? v : 0.f;
      ob[(size_t)o * N1_ + nHalf + t * 16 + n0] = v;
    }
  }
}

// ---------------------------------------------------------------------------
extern "C" void kernel_launch(void* const* d_in, const int* in_sizes, int n_in,
                              void* d_out, int out_size, void* d_ws,
                              size_t ws_size, hipStream_t stream) {
  (void)in_sizes; (void)n_in; (void)out_size; (void)ws_size;
  const float* xyzs     = (const float*)d_in[0];
  const float* oxyz     = (const float*)d_in[1];
  const float* features = (const float*)d_in[2];
  const float* ofeat    = (const float*)d_in[3];
  const float* Wt0      = (const float*)d_in[4];
  const float* Wt1      = (const float*)d_in[5];
  const float* Ws       = (const float*)d_in[6];
  const float* g        = (const float*)d_in[7];
  const float* be       = (const float*)d_in[8];
  const float* mn       = (const float*)d_in[9];
  const float* vr       = (const float*)d_in[10];
  float* out = (float*)d_out;

  // workspace layout (bf16): tt (B,L2,KT,CT,N2) then xbuf (B,L1,CCAT,N1)
  const size_t ttElems = (size_t)B_ * L2_ * KT_ * CT_ * N2_;   // 6,291,456
  bf16_t* tt   = (bf16_t*)d_ws;
  bf16_t* xbuf = (bf16_t*)((char*)d_ws + ttElems * sizeof(bf16_t));

  // output prefix: original_xyzs passthrough
  hipMemcpyAsync(out, oxyz, sizeof(float) * (size_t)B_ * L1_ * N1_ * 3,
                 hipMemcpyDeviceToDevice, stream);

  k_temporal_conv<<<B_ * L2_ * KT_ * (N2_ / 64), 256, 0, stream>>>(
      features, Wt0, Wt1, tt);
  k_knn_interp<<<B_ * L1_ * (N1_ / 256), 256, 0, stream>>>(
      xyzs, oxyz, ofeat, tt, xbuf);
  k_out_gemm<<<B_ * L1_ * (CS_ / 64) * (N1_ / 64), 256, 0, stream>>>(
      xbuf, Ws, g, be, mn, vr, out);
}
